// CapsModel_78340203479699
// MI455X (gfx1250) — matrix operand order repaired and checked by reading, not compiled, not run on hardware
//
#include <hip/hip_runtime.h>
#include <hip/hip_bf16.h>
#include <cstdint>

// ---------------------------------------------------------------------------
// CDNA5 / gfx1250 implementation of the multimodal capsule model.
// All matmuls (encoder LSTM input+recurrent projections, routing LSTMs,
// modality projections, fc1) run on v_wmma_f32_16x16x32_f16.
//
// Round-1 fix: the scan kernel previously fully unrolled the gate-tile loops,
// letting LICM hoist ~192 loop-invariant weight fragments out of the 50-step
// time loop -> >256 VGPRs -> scratch spills feeding every WMMA. Now the gate
// loop is kept rolled (#pragma unroll 1), K loops are unrolled only 2x, and
// the x tile is staged per-step into LDS as f16, so the register live set
// stays well under 256 and WMMA operands come from registers/LDS/cache.
// ---------------------------------------------------------------------------

typedef __attribute__((ext_vector_type(16))) _Float16 v16h;
typedef __attribute__((ext_vector_type(8)))  _Float16 v8h;
typedef __attribute__((ext_vector_type(8)))  float    v8f;

#define DEV_INLINE __device__ __forceinline__

DEV_INLINE float sigmoid_f(float x){ return 1.0f / (1.0f + __expf(-x)); }

DEV_INLINE v8f wmma16(v16h a, v16h b, v8f c){
  // D = A(16x32 f16) * B(32x16 f16) + C(16x16 f32)
  return __builtin_amdgcn_wmma_f32_16x16x32_f16(false, a, false, b, (short)0, c, false, false);
}

// A fragment (16x32 f16) from a row-major f32 matrix, with K bound (zero pad).
// Lane holds row (lane&15); K chunks [kc+koff, +8) and [kc+16+koff, +8), koff=8*(lane>=16).
DEV_INLINE v16h load_a_f32(const float* __restrict__ rowPtr, int kc, int Klim, int half_id){
  v16h a;
  const int koff = kc + half_id * 8;
#pragma unroll
  for (int j = 0; j < 8; ++j){ int k = koff + j;      a[j]     = (k < Klim) ? (_Float16)rowPtr[k] : (_Float16)0.0f; }
#pragma unroll
  for (int j = 0; j < 8; ++j){ int k = koff + 16 + j; a[8 + j] = (k < Klim) ? (_Float16)rowPtr[k] : (_Float16)0.0f; }
  return a;
}

// A fragment from an f16 row (LDS x/h tile), no bounds needed.
DEV_INLINE v16h load_a_h(const _Float16* rowPtr, int kc, int half_id){
  v8h lo = *(const v8h*)(rowPtr + kc + half_id * 8);
  v8h hi = *(const v8h*)(rowPtr + kc + 16 + half_id * 8);
  v16h a;
#pragma unroll
  for (int j = 0; j < 8; ++j){ a[j] = lo[j]; a[8 + j] = hi[j]; }
  return a;
}

// B fragment (32x16 f16) from prepped weights W[N][Kpad] (row = output col).
// Lane holds col (lane&15); K chunks [kc+koff, +8) and [kc+16+koff, +8), koff=8*(lane>=16).
DEV_INLINE v16h load_b_h(const _Float16* __restrict__ W, int Kpad, int ntile, int kc,
                         int l16, int half_id){
  const _Float16* p = W + (size_t)(ntile * 16 + l16) * Kpad + kc + half_id * 8;
  v8h lo = *(const v8h*)p;   // 16B aligned: Kpad multiple of 32 halfs
  v8h hi = *(const v8h*)(p + 16);
  v16h b;
#pragma unroll
  for (int j = 0; j < 8; ++j){ b[j] = lo[j]; b[8 + j] = hi[j]; }
  return b;
}

// ---------------------------------------------------------------------------
// Fused LSTM scan: gates = x_t @ Wih^T + h @ Whh^T + (bih+bhh), batch-parallel.
// Each wave owns 16 batch rows; c in registers (C/D layout), x and h tiles
// live in per-wave LDS regions (same-wave DS ops are in-order -> no barriers).
// ---------------------------------------------------------------------------
template<int INDIM, int HID>
__global__ void __launch_bounds__(128)
lstm_scan_kernel(const float* __restrict__ x, int Klim, int xRowStride, int xTStride,
                 const _Float16* __restrict__ Wih, const _Float16* __restrict__ Whh,
                 const float* __restrict__ bias,
                 int T, int t0, int tdir,
                 float* __restrict__ out, int outStride, int outColOff, int outMode,
                 int Btot)
{
  constexpr int NT  = HID / 16;   // tiles per gate type
  constexpr int KCX = INDIM / 32; // input K chunks
  constexpr int KCH = HID / 32;   // hidden K chunks
  __shared__ _Float16 ldsX[4][16][INDIM];
  __shared__ _Float16 ldsH[4][16][HID];

  const int wv      = threadIdx.x >> 5;
  const int lane    = threadIdx.x & 31;
  const int half_id = lane >> 4;
  const int l16     = lane & 15;
  const int row0    = (blockIdx.x * 4 + wv) * 16;
  if (row0 >= Btot) return;  // wave-uniform: EXEC stays all-ones for WMMA

  v8f c[NT];
#pragma unroll
  for (int n = 0; n < NT; ++n)
#pragma unroll
    for (int j = 0; j < 8; ++j) c[n][j] = 0.0f;

  v16h ha[KCH];
#pragma unroll
  for (int k = 0; k < KCH; ++k)
#pragma unroll
    for (int j = 0; j < 16; ++j) ha[k][j] = (_Float16)0.0f;

  const _Float16* xArow = &ldsX[wv][l16][0];   // A-frag source row for this lane

  for (int s = 0; s < T; ++s){
    const int t = t0 + s * tdir;
    const float* xbase = x + (size_t)row0 * xRowStride + (size_t)t * xTStride;

    // Stage x tile [16 x INDIM] into LDS as f16 (vectorized, guarded f32 reads).
#pragma unroll 1
    for (int idx = lane; idx < (16 * INDIM) / 8; idx += 32){
      const int r  = idx / (INDIM / 8);
      const int c8 = (idx - r * (INDIM / 8)) * 8;
      const float* xr = xbase + (size_t)r * xRowStride + c8;
      v8h h8;
#pragma unroll
      for (int j = 0; j < 8; ++j)
        h8[j] = (c8 + j < Klim) ? (_Float16)xr[j] : (_Float16)0.0f;
      *(v8h*)&ldsX[wv][r][c8] = h8;
    }

    // Gate-tile loop kept rolled: weight-fragment addresses depend on n, so
    // LICM cannot hoist them out of the time loop (prevents spill storm).
#pragma unroll 1
    for (int n = 0; n < NT; ++n){
      v8f acc[4];  // i, f, g, o tiles for this 16-col group
#pragma unroll
      for (int g = 0; g < 4; ++g){
        const int gt = n + g * NT;
        const float bv = bias[gt * 16 + l16];
        v8f a;
#pragma unroll
        for (int j = 0; j < 8; ++j) a[j] = bv;
#pragma unroll 2
        for (int k = 0; k < KCX; ++k)
          a = wmma16(load_a_h(xArow, k * 32, half_id),
                     load_b_h(Wih, INDIM, gt, k * 32, l16, half_id), a);
#pragma unroll 2
        for (int k = 0; k < KCH; ++k)
          a = wmma16(ha[k], load_b_h(Whh, HID, gt, k * 32, l16, half_id), a);
        acc[g] = a;
      }
#pragma unroll
      for (int j = 0; j < 8; ++j){
        float ig = sigmoid_f(acc[0][j]);
        float fg = sigmoid_f(acc[1][j]);
        float gg = tanhf(acc[2][j]);
        float og = sigmoid_f(acc[3][j]);
        float cj = fg * c[n][j] + ig * gg;
        c[n][j] = cj;
        ldsH[wv][j + half_id * 8][n * 16 + l16] = (_Float16)(og * tanhf(cj));
      }
    }
    // repack new h (D layout in LDS) into A fragments for next step
#pragma unroll
    for (int k = 0; k < KCH; ++k) ha[k] = load_a_h(&ldsH[wv][l16][0], k * 32, half_id);
  }

#pragma unroll 1
  for (int idx = lane; idx < 16 * HID; idx += 32){
    const int r = idx / HID, ck = idx % HID;
    float v = (float)ldsH[wv][r][ck];
    if (outMode >= 1) v = tanhf(v);
    float* dst = out + (size_t)(row0 + r) * outStride + outColOff + ck;
    if (outMode == 2) *dst += v; else *dst = v;
  }
}

// ---------------------------------------------------------------------------
// Simple WMMA GEMM: out[M,N] = act(A[M,K] * W^T + bias), W prepped f16 [N][K].
// One 16x16 tile per wave.
// ---------------------------------------------------------------------------
template<int K>
__global__ void __launch_bounds__(128)
gemm_wmma_kernel(const float* __restrict__ A, int lda,
                 const _Float16* __restrict__ W,
                 const float* __restrict__ bias,
                 float* __restrict__ out, int ldc,
                 int M, int N, int act)
{
  const int wv = threadIdx.x >> 5;
  const int lane = threadIdx.x & 31;
  const int half_id = lane >> 4, l16 = lane & 15;
  const int tilesN = N >> 4;
  const int wg = blockIdx.x * 4 + wv;
  const int mt = wg / tilesN, nt = wg - mt * tilesN;
  if (mt * 16 >= M) return;

  const float* arow = A + (size_t)(mt * 16 + l16) * lda;
  const float bv = bias ? bias[nt * 16 + l16] : 0.0f;
  v8f acc;
#pragma unroll
  for (int j = 0; j < 8; ++j) acc[j] = bv;
#pragma unroll 2
  for (int k = 0; k < K / 32; ++k){
    v16h a = load_a_f32(arow, k * 32, K, half_id);
    v16h b = load_b_h(W, K, nt, k * 32, l16, half_id);
    acc = wmma16(a, b, acc);
  }
#pragma unroll
  for (int j = 0; j < 8; ++j){
    float v = acc[j];
    if (act) v = tanhf(v);
    out[(size_t)(mt * 16 + j + half_id * 8) * ldc + nt * 16 + l16] = v;
  }
}

// ---------------------------------------------------------------------------
// Pointwise / glue kernels
// ---------------------------------------------------------------------------
__global__ void prep_w16_kernel(const float* __restrict__ src, _Float16* __restrict__ dst,
                                int N, int Ksrc, int Kpad, int transpose)
{
  int i = blockIdx.x * blockDim.x + threadIdx.x;
  if (i >= N * Kpad) return;
  int n = i / Kpad, k = i - n * Kpad;
  float v = 0.0f;
  if (k < Ksrc) v = transpose ? src[(size_t)k * N + n] : src[(size_t)n * Ksrc + k];
  dst[i] = (_Float16)v;
}

__global__ void add_bias_kernel(const float* a, const float* b, float* dst, int n){
  int i = blockIdx.x * blockDim.x + threadIdx.x;
  if (i < n) dst[i] = a[i] + b[i];
}

__global__ void fill_kernel(float* p, float v, int n){
  int i = blockIdx.x * blockDim.x + threadIdx.x;
  if (i < n) p[i] = v;
}

__global__ void copy_slot_kernel(float* dst, const float* src, int nslots, int slot, int Btot){
  int i = blockIdx.x * blockDim.x + threadIdx.x;
  if (i >= Btot * 128) return;
  int b = i >> 7, d = i & 127;
  dst[((size_t)b * nslots + slot) * 128 + d] = src[i];
}

__global__ void softmax_rc_kernel(float* rc, int N, int Btot){
  int b = blockIdx.x * blockDim.x + threadIdx.x;
  if (b >= Btot) return;
  float v[8]; float m = -1e30f;
  for (int n = 0; n < N; ++n){ v[n] = rc[b * N + n]; m = fmaxf(m, v[n]); }
  float s = 0.f;
  for (int n = 0; n < N; ++n){ v[n] = __expf(v[n] - m); s += v[n]; }
  float inv = 1.0f / s;
  for (int n = 0; n < N; ++n) rc[b * N + n] = v[n] * inv;
}

__global__ void mix_scale_kernel(float* __restrict__ xout, const float* __restrict__ caps,
                                 const float* __restrict__ rc, int N, int Btot){
  int i = blockIdx.x * blockDim.x + threadIdx.x;
  if (i >= Btot * N * 128) return;
  int bn = i >> 7;                  // b*N + n  (caps flat [B][N][128])
  xout[i] = rc[bn] * caps[i];
}

__global__ void agree_kernel(float* __restrict__ rc, const float* __restrict__ caps,
                             const float* __restrict__ parent,
                             int pStride, int pColOff, int N, int Btot){
  int i = blockIdx.x * blockDim.x + threadIdx.x;
  if (i >= Btot * N) return;
  int b = i / N;
  const float* cp = caps + (size_t)i * 128;
  const float* pp = parent + (size_t)b * pStride + pColOff;
  float s = 0.f;
  for (int d = 0; d < 128; ++d) s += cp[d] * pp[d];
  rc[i] += s;
}

__global__ void caps_write_kernel(float* __restrict__ capsOut, const float* __restrict__ deci,
                                  const float* __restrict__ dc, int Btot){
  int i = blockIdx.x * blockDim.x + threadIdx.x;
  if (i >= Btot * 8 * 128) return;
  int d = i & 127;
  int n = (i >> 7) & 7;
  int b = i >> 10;
  capsOut[i] = (n < 7) ? deci[((size_t)b * 7 + n) * 128 + d] : dc[(size_t)b * 128 + d];
}

__global__ void fc2_head_kernel(float* __restrict__ preds, const float* __restrict__ out1,
                                const float* __restrict__ w, const float* __restrict__ bb, int Btot){
  int b = blockIdx.x * blockDim.x + threadIdx.x;
  if (b >= Btot) return;
  float s = bb[0];
  for (int k = 0; k < 64; ++k) s += out1[(size_t)b * 64 + k] * w[k];
  preds[b] = s;
}

// ---------------------------------------------------------------------------
// Host orchestration (graph-capture safe: everything on `stream`, ws bump alloc)
// ---------------------------------------------------------------------------
extern "C" void kernel_launch(void* const* d_in, const int* in_sizes, int n_in,
                              void* d_out, int out_size, void* d_ws, size_t ws_size,
                              hipStream_t stream)
{
  (void)n_in; (void)out_size; (void)ws_size;
  constexpr int SEQ = 50, TIN = 300, AIN = 74, VIN = 35;
  const int B_ = in_sizes[0] / (SEQ * TIN);

  const float* text  = (const float*)d_in[0];
  const float* audio = (const float*)d_in[1];
  const float* video = (const float*)d_in[2];
  // d_in[3] = batch_size scalar (unused; derived from in_sizes)
#define PRM(i) ((const float*)d_in[4 + (i)])
  // Param leaf order:
  //  0..3 enc_t_f{Wih,Whh,bih,bhh}  4..7 enc_t_b  8..11 enc_a_f  12..15 enc_a_b
  // 16..19 enc_v_f 20..23 enc_v_b   24 Wt 25 Wa 26 Wv
  // 27..30 lstm_ta 31..34 lstm_tv 35..38 lstm_av 39..42 lstm_tav
  // 43..46 deci_f 47..50 deci_b    51 fc1_w 52 fc1_b 53 fc2_w 54 fc2_b

  char* wsBase = (char*)d_ws;
  size_t cur = 0;
  auto alloc = [&](size_t bytes) -> void* {
    cur = (cur + 255) & ~(size_t)255;
    void* p = wsBase + cur;
    cur += bytes;
    return p;
  };

  const int encIpad[3] = {320, 96, 64};
  const int encI[3]    = {TIN, AIN, VIN};
  _Float16 *encWih[6], *encWhh[6]; float* encB[6];
  for (int m = 0; m < 3; ++m)
    for (int dir = 0; dir < 2; ++dir){
      int e = m * 2 + dir;
      encWih[e] = (_Float16*)alloc((size_t)256 * encIpad[m] * 2);
      encWhh[e] = (_Float16*)alloc((size_t)256 * 64 * 2);
      encB[e]   = (float*)alloc(256 * 4);
    }
  _Float16 *rWih[6], *rWhh[6]; float* rB[6];   // ta, tv, av, tav, deci_f, deci_b
  for (int e = 0; e < 6; ++e){
    rWih[e] = (_Float16*)alloc((size_t)512 * 128 * 2);
    rWhh[e] = (_Float16*)alloc((size_t)512 * 128 * 2);
    rB[e]   = (float*)alloc(512 * 4);
  }
  _Float16* projP[3];
  for (int m = 0; m < 3; ++m) projP[m] = (_Float16*)alloc((size_t)4 * 128 * 128 * 2);
  _Float16* fc1p = (_Float16*)alloc((size_t)64 * 128 * 2);

  float* tvec  = (float*)alloc((size_t)B_ * 128 * 4);
  float* avec  = (float*)alloc((size_t)B_ * 128 * 4);
  float* vvec  = (float*)alloc((size_t)B_ * 128 * 4);
  float* tusc  = (float*)alloc((size_t)4 * B_ * 128 * 4);
  float* ausc  = (float*)alloc((size_t)4 * B_ * 128 * 4);
  float* vusc  = (float*)alloc((size_t)4 * B_ * 128 * 4);
  float* ta_c  = (float*)alloc((size_t)B_ * 2 * 128 * 4);
  float* tv_c  = (float*)alloc((size_t)B_ * 2 * 128 * 4);
  float* av_c  = (float*)alloc((size_t)B_ * 2 * 128 * 4);
  float* tav_c = (float*)alloc((size_t)B_ * 3 * 128 * 4);
  float* deci  = (float*)alloc((size_t)B_ * 7 * 128 * 4);
  float* dc    = (float*)alloc((size_t)B_ * 128 * 4);
  float* xmix  = (float*)alloc((size_t)B_ * 7 * 128 * 4);
  float* rc0   = (float*)alloc((size_t)B_ * 2 * 4);
  float* rc1   = (float*)alloc((size_t)B_ * 2 * 4);
  float* rc2   = (float*)alloc((size_t)B_ * 2 * 4);
  float* rc3   = (float*)alloc((size_t)B_ * 3 * 4);
  float* rc4   = (float*)alloc((size_t)B_ * 7 * 4);
  float* out1  = (float*)alloc((size_t)B_ * 64 * 4);

  auto pwGrid = [](long n){ return dim3((unsigned)((n + 255) / 256)); };

  // ---- weight prep (f32 -> f16 [N][Kpad], bias combine) ----
  for (int m = 0; m < 3; ++m)
    for (int dir = 0; dir < 2; ++dir){
      int e = m * 2 + dir, base = 4 * e;
      prep_w16_kernel<<<pwGrid((long)256 * encIpad[m]), 256, 0, stream>>>(PRM(base + 0), encWih[e], 256, encI[m], encIpad[m], 0);
      prep_w16_kernel<<<pwGrid(256 * 64), 256, 0, stream>>>(PRM(base + 1), encWhh[e], 256, 64, 64, 0);
      add_bias_kernel<<<pwGrid(256), 256, 0, stream>>>(PRM(base + 2), PRM(base + 3), encB[e], 256);
    }
  for (int e = 0; e < 6; ++e){
    int base = 27 + 4 * e;
    prep_w16_kernel<<<pwGrid(512 * 128), 256, 0, stream>>>(PRM(base + 0), rWih[e], 512, 128, 128, 0);
    prep_w16_kernel<<<pwGrid(512 * 128), 256, 0, stream>>>(PRM(base + 1), rWhh[e], 512, 128, 128, 0);
    add_bias_kernel<<<pwGrid(512), 256, 0, stream>>>(PRM(base + 2), PRM(base + 3), rB[e], 512);
  }
  for (int m = 0; m < 3; ++m)
    for (int i = 0; i < 4; ++i)
      prep_w16_kernel<<<pwGrid(128 * 128), 256, 0, stream>>>(PRM(24 + m) + (size_t)i * 128 * 128,
                                                             projP[m] + (size_t)i * 128 * 128, 128, 128, 128, 1);
  prep_w16_kernel<<<pwGrid(64 * 128), 256, 0, stream>>>(PRM(51), fc1p, 64, 128, 128, 0);

  // ---- encoder BiLSTMs (fused input projection + recurrence) ----
  const int scanBlocks = (B_ + 63) / 64;
  lstm_scan_kernel<320, 64><<<scanBlocks, 128, 0, stream>>>(text, TIN, SEQ * TIN, TIN, encWih[0], encWhh[0], encB[0], SEQ, 0, 1, tvec, 128, 0, 1, B_);
  lstm_scan_kernel<320, 64><<<scanBlocks, 128, 0, stream>>>(text, TIN, SEQ * TIN, TIN, encWih[1], encWhh[1], encB[1], SEQ, SEQ - 1, -1, tvec, 128, 64, 1, B_);
  lstm_scan_kernel< 96, 64><<<scanBlocks, 128, 0, stream>>>(audio, AIN, SEQ * AIN, AIN, encWih[2], encWhh[2], encB[2], SEQ, 0, 1, avec, 128, 0, 1, B_);
  lstm_scan_kernel< 96, 64><<<scanBlocks, 128, 0, stream>>>(audio, AIN, SEQ * AIN, AIN, encWih[3], encWhh[3], encB[3], SEQ, SEQ - 1, -1, avec, 128, 64, 1, B_);
  lstm_scan_kernel< 64, 64><<<scanBlocks, 128, 0, stream>>>(video, VIN, SEQ * VIN, VIN, encWih[4], encWhh[4], encB[4], SEQ, 0, 1, vvec, 128, 0, 1, B_);
  lstm_scan_kernel< 64, 64><<<scanBlocks, 128, 0, stream>>>(video, VIN, SEQ * VIN, VIN, encWih[5], encWhh[5], encB[5], SEQ, SEQ - 1, -1, vvec, 128, 64, 1, B_);

  // ---- 12 modality projections (out[i] = vec @ W[i]) ----
  const int gemmBlocks128 = ((B_ / 16) * 8 + 3) / 4;
  float* vecs[3] = {tvec, avec, vvec};
  float* uscs[3] = {tusc, ausc, vusc};
  for (int m = 0; m < 3; ++m)
    for (int i = 0; i < 4; ++i)
      gemm_wmma_kernel<128><<<gemmBlocks128, 128, 0, stream>>>(vecs[m], 128, projP[m] + (size_t)i * 128 * 128,
                                                               nullptr, uscs[m] + (size_t)i * B_ * 128, 128, B_, 128, 0);

  // ---- capsule stacks ----
  auto slotCopy = [&](float* dst, const float* src, int nslots, int slot){
    copy_slot_kernel<<<pwGrid((long)B_ * 128), 256, 0, stream>>>(dst, src, nslots, slot, B_);
  };
  slotCopy(ta_c,  tusc + 0 * (size_t)B_ * 128, 2, 0);
  slotCopy(ta_c,  ausc + 0 * (size_t)B_ * 128, 2, 1);
  slotCopy(tv_c,  tusc + 1 * (size_t)B_ * 128, 2, 0);
  slotCopy(tv_c,  vusc + 0 * (size_t)B_ * 128, 2, 1);
  slotCopy(av_c,  ausc + 1 * (size_t)B_ * 128, 2, 0);
  slotCopy(av_c,  vusc + 1 * (size_t)B_ * 128, 2, 1);
  slotCopy(tav_c, tusc + 2 * (size_t)B_ * 128, 3, 0);
  slotCopy(tav_c, ausc + 2 * (size_t)B_ * 128, 3, 1);
  slotCopy(tav_c, vusc + 2 * (size_t)B_ * 128, 3, 2);
  slotCopy(deci,  tusc + 3 * (size_t)B_ * 128, 7, 0);
  slotCopy(deci,  ausc + 3 * (size_t)B_ * 128, 7, 1);
  slotCopy(deci,  vusc + 3 * (size_t)B_ * 128, 7, 2);

  fill_kernel<<<pwGrid(B_ * 2), 256, 0, stream>>>(rc0, 1.0f, B_ * 2);
  fill_kernel<<<pwGrid(B_ * 2), 256, 0, stream>>>(rc1, 1.0f, B_ * 2);
  fill_kernel<<<pwGrid(B_ * 2), 256, 0, stream>>>(rc2, 1.0f, B_ * 2);
  fill_kernel<<<pwGrid(B_ * 3), 256, 0, stream>>>(rc3, 1.0f, B_ * 3);
  fill_kernel<<<pwGrid(B_ * 7), 256, 0, stream>>>(rc4, 1.0f, B_ * 7);

  // ---- routing iterations ----
  float* capsOut = (float*)d_out + B_;   // caps[4][B][8][128] after preds[B]
  const float* capsArr[4] = {ta_c, tv_c, av_c, tav_c};
  float* rcArr[4] = {rc0, rc1, rc2, rc3};
  const int Narr[4] = {2, 2, 2, 3};

  for (int r = 0; r < 4; ++r){
    softmax_rc_kernel<<<pwGrid(B_), 256, 0, stream>>>(rc0, 2, B_);
    softmax_rc_kernel<<<pwGrid(B_), 256, 0, stream>>>(rc1, 2, B_);
    softmax_rc_kernel<<<pwGrid(B_), 256, 0, stream>>>(rc2, 2, B_);
    softmax_rc_kernel<<<pwGrid(B_), 256, 0, stream>>>(rc3, 3, B_);
    softmax_rc_kernel<<<pwGrid(B_), 256, 0, stream>>>(rc4, 7, B_);

    for (int q = 0; q < 4; ++q){
      mix_scale_kernel<<<pwGrid((long)B_ * Narr[q] * 128), 256, 0, stream>>>(xmix, capsArr[q], rcArr[q], Narr[q], B_);
      lstm_scan_kernel<128, 128><<<scanBlocks, 128, 0, stream>>>(xmix, 128, Narr[q] * 128, 128,
          rWih[q], rWhh[q], rB[q], Narr[q], 0, 1, deci, 7 * 128, (3 + q) * 128, 1, B_);
    }
    mix_scale_kernel<<<pwGrid((long)B_ * 7 * 128), 256, 0, stream>>>(xmix, deci, rc4, 7, B_);
    lstm_scan_kernel<128, 128><<<scanBlocks, 128, 0, stream>>>(xmix, 128, 7 * 128, 128,
        rWih[4], rWhh[4], rB[4], 7, 0, 1, dc, 128, 0, 1, B_);
    lstm_scan_kernel<128, 128><<<scanBlocks, 128, 0, stream>>>(xmix, 128, 7 * 128, 128,
        rWih[5], rWhh[5], rB[5], 7, 6, -1, dc, 128, 0, 2, B_);

    caps_write_kernel<<<pwGrid((long)B_ * 8 * 128), 256, 0, stream>>>(capsOut + (size_t)r * B_ * 8 * 128, deci, dc, B_);

    if (r < 3){
      agree_kernel<<<pwGrid(B_ * 2), 256, 0, stream>>>(rc0, ta_c,  deci, 7 * 128, 3 * 128, 2, B_);
      agree_kernel<<<pwGrid(B_ * 2), 256, 0, stream>>>(rc1, tv_c,  deci, 7 * 128, 4 * 128, 2, B_);
      agree_kernel<<<pwGrid(B_ * 2), 256, 0, stream>>>(rc2, av_c,  deci, 7 * 128, 5 * 128, 2, B_);
      agree_kernel<<<pwGrid(B_ * 3), 256, 0, stream>>>(rc3, tav_c, deci, 7 * 128, 6 * 128, 3, B_);
      agree_kernel<<<pwGrid(B_ * 7), 256, 0, stream>>>(rc4, deci,  dc,   128,     0,       7, B_);
    }
  }

  // ---- head ----
  const int fc1Blocks = ((B_ / 16) * 4 + 3) / 4;
  gemm_wmma_kernel<128><<<fc1Blocks, 128, 0, stream>>>(dc, 128, fc1p, PRM(52), out1, 64, B_, 64, 1);
  fc2_head_kernel<<<pwGrid(B_), 256, 0, stream>>>((float*)d_out, out1, PRM(53), PRM(54), B_);
#undef PRM
}
